// LocalAttention_25769804308
// MI455X (gfx1250) — compile-verified, matrix operand order checked by reading, not run on hardware
//
#include <hip/hip_runtime.h>

#define D_MODEL 512
#define N_HEADS 8
#define D_HEAD  64
#define DFF     2048
#define WIN     128
#define SEQLEN  1024
#define NBATCH  8
#define NROWS   (SEQLEN * NBATCH)   // 8192 token rows, row = s*NBATCH + n

typedef float  v8f   __attribute__((ext_vector_type(8)));
typedef __bf16 v16bf __attribute__((ext_vector_type(16)));
typedef __bf16 bf16x2 __attribute__((ext_vector_type(2)));
typedef unsigned int u32x4 __attribute__((ext_vector_type(4)));
typedef unsigned int u32x2 __attribute__((ext_vector_type(2)));
typedef float  f32x4 __attribute__((ext_vector_type(4)));

struct FragBits { u32x4 lo, hi; };
struct U32x8    { unsigned int w[8]; };

// packed fp32x2 -> bf16x2 (RNE); should lower to v_cvt_pk_bf16_f32 on gfx1250
__device__ __forceinline__ unsigned int cvtpk_bf16(float x, float y) {
  bf16x2 v;
  v[0] = (__bf16)x;
  v[1] = (__bf16)y;
  return __builtin_bit_cast(unsigned int, v);
}

__device__ __forceinline__ unsigned short f2bf1(float f) {
  return __builtin_bit_cast(unsigned short, (__bf16)f);
}

__device__ __forceinline__ u32x2 pack4bf(f32x4 f) {
  u32x2 r;
  r.x = cvtpk_bf16(f.x, f.y);
  r.y = cvtpk_bf16(f.z, f.w);
  return r;
}

// 16x32 bf16 operand fragment from LDS (rows = M (A) or N (B), cols = K).
// Matches CDNA5 16-bit A layout: lanes 0-15 hold K 0..7 / 16..23,
// lanes 16-31 hold K 8..15 / 24..31 (two 16B ds loads per lane).
__device__ __forceinline__ v16bf load_frag_bf(const unsigned short* base, int stride, int lane) {
  const int r = lane & 15, kh = lane >> 4;
  const unsigned short* p = base + r * stride + kh * 8;
  FragBits fb;
  fb.lo = *(const u32x4*)(p);
  fb.hi = *(const u32x4*)(p + 16);
  return __builtin_bit_cast(v16bf, fb);
}

// Same fragment, but source is fp32 in LDS (softmax probabilities) -> cvt to bf16
__device__ __forceinline__ v16bf load_frag_f32(const float* base, int stride, int lane) {
  const int r = lane & 15, kh = lane >> 4;
  const float* p = base + r * stride + kh * 8;
  f32x4 f0 = *(const f32x4*)(p);
  f32x4 f1 = *(const f32x4*)(p + 4);
  f32x4 f2 = *(const f32x4*)(p + 16);
  f32x4 f3 = *(const f32x4*)(p + 20);
  U32x8 t;
  t.w[0] = cvtpk_bf16(f0.x, f0.y); t.w[1] = cvtpk_bf16(f0.z, f0.w);
  t.w[2] = cvtpk_bf16(f1.x, f1.y); t.w[3] = cvtpk_bf16(f1.z, f1.w);
  t.w[4] = cvtpk_bf16(f2.x, f2.y); t.w[5] = cvtpk_bf16(f2.z, f2.w);
  t.w[6] = cvtpk_bf16(f3.x, f3.y); t.w[7] = cvtpk_bf16(f3.z, f3.w);
  return __builtin_bit_cast(v16bf, t);
}

__device__ __forceinline__ v8f wmma_bf16(v16bf a, v16bf b, v8f c) {
  // (neg_a, A, neg_b, B, c_mod, C, reuse_a, reuse_b)
  return __builtin_amdgcn_wmma_f32_16x16x32_bf16(false, a, false, b, (short)0, c, false, false);
}

// ---------------------------------------------------------------------------
// GEMM:  out[m,n] = sum_k A[m,k] * W[n,k] + bias[n]  (+ residual[m,n]) (+relu)
// A:[M,K] lda, W:[N,K] ldw (torch Linear weight). 128x128 tile, K-step 32,
// 256 threads = 8 wave32 waves in a 4x2 grid; each wave owns 32x64 of C
// (2x4 v_wmma_f32_16x16x32_bf16 accumulators). Double-buffered LDS: one
// barrier per K-step, next tile's global loads issued before current tile's
// fragment loads so VMEM latency hides under WMMA.
// ---------------------------------------------------------------------------
#define LDSS 40   // 32 K + 8 pad (80B row stride: 16B aligned, conflict-free)

template <bool HAS_RES, bool RELU>
__global__ __launch_bounds__(256)
void gemm_bf16_kernel(const float* __restrict__ A, int lda,
                      const float* __restrict__ W, int ldw,
                      const float* __restrict__ bias,
                      const float* __restrict__ residual, int ldr,
                      float* __restrict__ out, int ldo,
                      int K)
{
  __shared__ unsigned short As[2][128 * LDSS];
  __shared__ unsigned short Bs[2][128 * LDSS];
  const int tid  = threadIdx.x;
  const int lane = tid & 31, wid = tid >> 5;
  const int wm = wid & 3, wn = wid >> 2;
  const int m0 = blockIdx.x * 128;
  const int n0 = blockIdx.y * 128;

  // per-thread tile-load coordinates (1024 float4 per 128x32 tile, 4 each)
  int lr[4], lc[4];
#pragma unroll
  for (int i = 0; i < 4; ++i) {
    const int v = tid + i * 256;
    lr[i] = v >> 3;
    lc[i] = (v & 7) << 2;
  }

  v8f acc[2][4];
  {
    v8f z = {};
#pragma unroll
    for (int i = 0; i < 2; ++i)
#pragma unroll
      for (int j = 0; j < 4; ++j) acc[i][j] = z;
  }

  // preload tile 0 into buffer 0
#pragma unroll
  for (int i = 0; i < 4; ++i) {
    f32x4 fa = *(const f32x4*)(A + (size_t)(m0 + lr[i]) * lda + lc[i]);
    *(u32x2*)(&As[0][lr[i] * LDSS + lc[i]]) = pack4bf(fa);
    f32x4 fw = *(const f32x4*)(W + (size_t)(n0 + lr[i]) * ldw + lc[i]);
    *(u32x2*)(&Bs[0][lr[i] * LDSS + lc[i]]) = pack4bf(fw);
  }

  int buf = 0;
  for (int k0 = 0; k0 < K; k0 += 32) {
    __syncthreads();                       // tile `buf` visible to all waves
    if (k0 + 32 < K) {                     // issue next tile's loads first
      const int kn = k0 + 32;
      const int nb = buf ^ 1;
#pragma unroll
      for (int i = 0; i < 4; ++i) {
        f32x4 fa = *(const f32x4*)(A + (size_t)(m0 + lr[i]) * lda + kn + lc[i]);
        *(u32x2*)(&As[nb][lr[i] * LDSS + lc[i]]) = pack4bf(fa);
        f32x4 fw = *(const f32x4*)(W + (size_t)(n0 + lr[i]) * ldw + kn + lc[i]);
        *(u32x2*)(&Bs[nb][lr[i] * LDSS + lc[i]]) = pack4bf(fw);
      }
    }
    v16bf af[2], bw[4];
#pragma unroll
    for (int mi = 0; mi < 2; ++mi)
      af[mi] = load_frag_bf(&As[buf][(wm * 32 + mi * 16) * LDSS], LDSS, lane);
#pragma unroll
    for (int ni = 0; ni < 4; ++ni)
      bw[ni] = load_frag_bf(&Bs[buf][(wn * 64 + ni * 16) * LDSS], LDSS, lane);
#pragma unroll
    for (int mi = 0; mi < 2; ++mi)
#pragma unroll
      for (int ni = 0; ni < 4; ++ni)
        acc[mi][ni] = wmma_bf16(af[mi], bw[ni], acc[mi][ni]);
    buf ^= 1;
  }

  // Epilogue. C layout: VGPR r -> (M=r, lanes 0-15) / (M=r+8, lanes 16-31), N=lane&15
  const int halfsel = (lane >> 4) << 3;
#pragma unroll
  for (int ni = 0; ni < 4; ++ni) {
    const int n  = n0 + wn * 64 + ni * 16 + (lane & 15);
    const float bn = bias[n];
#pragma unroll
    for (int mi = 0; mi < 2; ++mi) {
      const int mbase = m0 + wm * 32 + mi * 16 + halfsel;
#pragma unroll
      for (int r = 0; r < 8; ++r) {
        const int m = mbase + r;
        float v = acc[mi][ni][r] + bn;
        if (HAS_RES) v += residual[(size_t)m * ldr + n];
        if (RELU)    v = fmaxf(v, 0.0f);
        out[(size_t)m * ldo + n] = v;
      }
    }
  }
}

// ---------------------------------------------------------------------------
// LayerNorm over D_MODEL=512, one row per 128-thread block (4 waves)
// ---------------------------------------------------------------------------
__global__ __launch_bounds__(128)
void layernorm_kernel(const float* __restrict__ x,
                      const float* __restrict__ g,
                      const float* __restrict__ b,
                      float* __restrict__ out)
{
  const int row = blockIdx.x;
  const int tid = threadIdx.x;
  const int lane = tid & 31, wid = tid >> 5;
  const float* xr = x + (size_t)row * D_MODEL;
  float v[4], s = 0.f, ss = 0.f;
#pragma unroll
  for (int i = 0; i < 4; ++i) {
    v[i] = xr[tid + i * 128];
    s += v[i]; ss += v[i] * v[i];
  }
#pragma unroll
  for (int off = 16; off > 0; off >>= 1) {
    s  += __shfl_xor(s,  off, 32);
    ss += __shfl_xor(ss, off, 32);
  }
  __shared__ float rs[4], rss[4];
  if (lane == 0) { rs[wid] = s; rss[wid] = ss; }
  __syncthreads();
  s  = rs[0]  + rs[1]  + rs[2]  + rs[3];
  ss = rss[0] + rss[1] + rss[2] + rss[3];
  const float mu  = s * (1.0f / D_MODEL);
  const float var = ss * (1.0f / D_MODEL) - mu * mu;
  const float inv = rsqrtf(var + 1e-5f);
  float* orow = out + (size_t)row * D_MODEL;
#pragma unroll
  for (int i = 0; i < 4; ++i) {
    const int c = tid + i * 128;
    orow[c] = (v[i] - mu) * inv * g[c] + b[c];
  }
}

// ---------------------------------------------------------------------------
// Attention: one workgroup (128 thr = 4 waves) = 64 query rows of one (n,h).
// Keeps full score strip for the block in LDS (uses CDNA5's 320KB WGP LDS):
//   scores[64][T+8] f32, Q[64][72] bf16, Ktile[32][72] bf16, Vt[64][40] bf16.
// Stage 1: S = scale*Q·Kᵀ via WMMA (banded range if is_local)
// Stage 2: row softmax in LDS (2 threads / row)
// Stage 3: O = P·V via WMMA (P converted f32->bf16 at fragment build)
// ---------------------------------------------------------------------------
__global__ __launch_bounds__(128)
void attention_kernel(const float* __restrict__ qbuf, int ldq,
                      const float* __restrict__ kbuf,
                      const float* __restrict__ vbuf, int ldkv,
                      float* __restrict__ out, int ldo,
                      int T, int is_local)
{
  extern __shared__ char smem_raw[];
  const int SLD = T + 8;
  float*          scores = (float*)smem_raw;
  unsigned short* Qs = (unsigned short*)(smem_raw + (size_t)64 * SLD * 4);
  unsigned short* Ks = Qs + 64 * 72;
  unsigned short* Vs = Ks + 32 * 72;   // transposed: [d][t_local], stride 40
  float*          red = (float*)(Vs + 64 * 40);

  const int tid  = threadIdx.x;
  const int lane = tid & 31, wid = tid >> 5;
  const int q0 = blockIdx.x * 64;
  const int n  = blockIdx.y >> 3;      // batch index
  const int h  = blockIdx.y & 7;       // head index
  const int hc = h * D_HEAD;
  const int halfsel = (lane >> 4) << 3;

  // Q block 64x64 -> LDS bf16
#pragma unroll
  for (int i = 0; i < 8; ++i) {
    int v = tid + i * 128;
    int r = v >> 4;
    int c4 = (v & 15) << 2;
    f32x4 f = *(const f32x4*)(qbuf + ((size_t)(q0 + r) * NBATCH + n) * ldq + hc + c4);
    *(u32x2*)(&Qs[r * 72 + c4]) = pack4bf(f);
  }
  int jlo = 0, jhi = T;
  if (is_local) {                       // band for the whole 64-row block
    jlo = q0 - WIN; if (jlo < 0) jlo = 0;
    jhi = q0 + 64 + WIN; if (jhi > T) jhi = T;
  }
  __syncthreads();

  const int qrow = wid * 16;            // this wave's 16 query rows (in-block)
  const v16bf qa0 = load_frag_bf(&Qs[qrow * 72], 72, lane);       // d 0..31
  const v16bf qa1 = load_frag_bf(&Qs[qrow * 72 + 32], 72, lane);  // d 32..63
  const float scale = 0.125f;           // 1/sqrt(64)

  // ---- scores = scale * Q Kᵀ (+ band mask) ----
  for (int t0 = jlo; t0 < jhi; t0 += 32) {
    __syncthreads();
#pragma unroll
    for (int i = 0; i < 4; ++i) {       // K tile 32x64 -> LDS bf16
      int v = tid + i * 128;
      int r = v >> 4;
      int c4 = (v & 15) << 2;
      f32x4 f = *(const f32x4*)(kbuf + ((size_t)(t0 + r) * NBATCH + n) * ldkv + hc + c4);
      *(u32x2*)(&Ks[r * 72 + c4]) = pack4bf(f);
    }
    __syncthreads();
#pragma unroll
    for (int nt = 0; nt < 2; ++nt) {
      v16bf kb0 = load_frag_bf(&Ks[nt * 16 * 72], 72, lane);
      v16bf kb1 = load_frag_bf(&Ks[nt * 16 * 72 + 32], 72, lane);
      v8f c = {};
      c = wmma_bf16(qa0, kb0, c);
      c = wmma_bf16(qa1, kb1, c);
      const int j = t0 + nt * 16 + (lane & 15);
#pragma unroll
      for (int r = 0; r < 8; ++r) {
        const int rloc = qrow + halfsel + r;
        const int m = q0 + rloc;
        float sc = c[r] * scale;
        if (is_local && (j < m - WIN || j > m + WIN)) sc = -__builtin_inff();
        scores[(size_t)rloc * SLD + j] = sc;
      }
    }
  }
  __syncthreads();

  // ---- softmax over [jlo, jhi), 2 threads per row ----
  {
    const int row = tid >> 1, half = tid & 1;
    const int mid = (jlo + jhi) >> 1;
    const int a0 = half ? mid : jlo;
    const int a1 = half ? jhi : mid;
    float* srow = scores + (size_t)row * SLD;
    float mx = -__builtin_inff();
    for (int j = a0; j < a1; ++j) mx = fmaxf(mx, srow[j]);
    red[tid] = mx;
    __syncthreads();
    const float rowmax = fmaxf(red[row * 2], red[row * 2 + 1]);
    __syncthreads();
    float sum = 0.f;
    for (int j = a0; j < a1; ++j) {
      float e = __expf(srow[j] - rowmax);
      srow[j] = e;
      sum += e;
    }
    red[tid] = sum;
    __syncthreads();
    const float inv = 1.0f / (red[row * 2] + red[row * 2 + 1]);
    for (int j = a0; j < a1; ++j) srow[j] *= inv;
  }
  __syncthreads();

  // ---- O = P V ----
  v8f o[4];
  { v8f z = {}; o[0] = z; o[1] = z; o[2] = z; o[3] = z; }
  for (int t0 = jlo; t0 < jhi; t0 += 32) {
    __syncthreads();
#pragma unroll
    for (int i = 0; i < 16; ++i) {      // V tile, stored transposed [d][t]
      int v = tid + i * 128;
      int tl = v >> 6;
      int d  = v & 63;
      float f = vbuf[((size_t)(t0 + tl) * NBATCH + n) * ldkv + hc + d];
      Vs[d * 40 + tl] = f2bf1(f);
    }
    __syncthreads();
    v16bf p = load_frag_f32(scores + (size_t)qrow * SLD + t0, SLD, lane);
#pragma unroll
    for (int ni = 0; ni < 4; ++ni) {
      v16bf vb = load_frag_bf(&Vs[ni * 16 * 40], 40, lane);
      o[ni] = wmma_bf16(p, vb, o[ni]);
    }
  }
#pragma unroll
  for (int ni = 0; ni < 4; ++ni) {
    const int d = ni * 16 + (lane & 15);
#pragma unroll
    for (int r = 0; r < 8; ++r) {
      const int m = q0 + qrow + halfsel + r;
      out[((size_t)m * NBATCH + n) * ldo + hc + d] = o[ni][r];
    }
  }
}

// ---------------------------------------------------------------------------
// Host orchestration. d_in order = [src, tgt] + jax pytree leaves of params
// (dicts flattened with sorted keys).
// ---------------------------------------------------------------------------
extern "C" void kernel_launch(void* const* d_in, const int* in_sizes, int n_in,
                              void* d_out, int out_size, void* d_ws, size_t ws_size,
                              hipStream_t stream)
{
  (void)in_sizes; (void)n_in; (void)out_size; (void)ws_size;
  const float* src = (const float*)d_in[0];
  const float* tgt = (const float*)d_in[1];
  // dec (sorted): W1,W2,Wo_ca,Wo_sa,Wqkv_ca,Wqkv_sa,b1,b2,bo_ca,bo_sa,bqkv_ca,
  //               bqkv_sa,ln1_b,ln1_g,ln2_b,ln2_g,ln3_b,ln3_g
  const float* dW1      = (const float*)d_in[2];
  const float* dW2      = (const float*)d_in[3];
  const float* dWo_ca   = (const float*)d_in[4];
  const float* dWo_sa   = (const float*)d_in[5];
  const float* dWqkv_ca = (const float*)d_in[6];
  const float* dWqkv_sa = (const float*)d_in[7];
  const float* db1      = (const float*)d_in[8];
  const float* db2      = (const float*)d_in[9];
  const float* dbo_ca   = (const float*)d_in[10];
  const float* dbo_sa   = (const float*)d_in[11];
  const float* dbqkv_ca = (const float*)d_in[12];
  const float* dbqkv_sa = (const float*)d_in[13];
  const float* dln1_b = (const float*)d_in[14];
  const float* dln1_g = (const float*)d_in[15];
  const float* dln2_b = (const float*)d_in[16];
  const float* dln2_g = (const float*)d_in[17];
  const float* dln3_b = (const float*)d_in[18];
  const float* dln3_g = (const float*)d_in[19];
  const float* dec_norm_b = (const float*)d_in[20];
  const float* dec_norm_g = (const float*)d_in[21];
  // enc (sorted): W1,W2,Wo,Wqkv,b1,b2,bo,bqkv,ln1_b,ln1_g,ln2_b,ln2_g
  const float* eW1   = (const float*)d_in[22];
  const float* eW2   = (const float*)d_in[23];
  const float* eWo   = (const float*)d_in[24];
  const float* eWqkv = (const float*)d_in[25];
  const float* eb1   = (const float*)d_in[26];
  const float* eb2   = (const float*)d_in[27];
  const float* ebo   = (const float*)d_in[28];
  const float* ebqkv = (const float*)d_in[29];
  const float* eln1_b = (const float*)d_in[30];
  const float* eln1_g = (const float*)d_in[31];
  const float* eln2_b = (const float*)d_in[32];
  const float* eln2_g = (const float*)d_in[33];
  const float* enc_norm_b = (const float*)d_in[34];
  const float* enc_norm_g = (const float*)d_in[35];

  // workspace carve-up (fp32)
  float* X    = (float*)d_ws;                       // 8192 x 512
  float* Tb   = X    + (size_t)NROWS * D_MODEL;     // 8192 x 512
  float* QKV  = Tb   + (size_t)NROWS * D_MODEL;     // 8192 x 1536
  float* ATTN = QKV  + (size_t)NROWS * 3 * D_MODEL; // 8192 x 512
  float* FF   = ATTN + (size_t)NROWS * D_MODEL;     // 8192 x 2048
  float* MEM  = FF   + (size_t)NROWS * DFF;         // 8192 x 512

  const size_t att_smem = (size_t)64 * (SEQLEN + 8) * 4
                        + (size_t)(64 * 72 + 32 * 72 + 64 * 40) * 2
                        + 128 * 4;                  // ~277 KB of the 320 KB WGP LDS

  auto gemm = [&](const float* A, int lda, const float* W, int ldw,
                  const float* bias, const float* res,
                  float* O, int ldo, int N, int K, int relu) {
    dim3 grid(NROWS / 128, N / 128);
    if (res)
      gemm_bf16_kernel<true, false><<<grid, dim3(256), 0, stream>>>(
          A, lda, W, ldw, bias, res, D_MODEL, O, ldo, K);
    else if (relu)
      gemm_bf16_kernel<false, true><<<grid, dim3(256), 0, stream>>>(
          A, lda, W, ldw, bias, res, D_MODEL, O, ldo, K);
    else
      gemm_bf16_kernel<false, false><<<grid, dim3(256), 0, stream>>>(
          A, lda, W, ldw, bias, res, D_MODEL, O, ldo, K);
  };
  auto lnorm = [&](const float* x, const float* g, const float* b, float* o) {
    layernorm_kernel<<<dim3(NROWS), dim3(128), 0, stream>>>(x, g, b, o);
  };
  auto attn = [&](const float* q, int ldq, const float* k, const float* v, int ldkv,
                  float* o, int is_local) {
    dim3 grid(SEQLEN / 64, NBATCH * N_HEADS);
    attention_kernel<<<grid, dim3(128), att_smem, stream>>>(q, ldq, k, v, ldkv,
                                                            o, D_MODEL, SEQLEN, is_local);
  };

  // ---------------- encoder ----------------
  const float* x = src;
  for (int l = 0; l < 6; ++l) {
    gemm(x, D_MODEL, eWqkv + (size_t)l * 3 * D_MODEL * D_MODEL, D_MODEL,
         ebqkv + (size_t)l * 3 * D_MODEL, nullptr, QKV, 3 * D_MODEL, 3 * D_MODEL, D_MODEL, 0);
    attn(QKV, 3 * D_MODEL, QKV + D_MODEL, QKV + 2 * D_MODEL, 3 * D_MODEL, ATTN, 1);
    gemm(ATTN, D_MODEL, eWo + (size_t)l * D_MODEL * D_MODEL, D_MODEL,
         ebo + (size_t)l * D_MODEL, x, Tb, D_MODEL, D_MODEL, D_MODEL, 0);
    lnorm(Tb, eln1_g + (size_t)l * D_MODEL, eln1_b + (size_t)l * D_MODEL, X);
    gemm(X, D_MODEL, eW1 + (size_t)l * DFF * D_MODEL, D_MODEL,
         eb1 + (size_t)l * DFF, nullptr, FF, DFF, DFF, D_MODEL, 1);
    gemm(FF, DFF, eW2 + (size_t)l * D_MODEL * DFF, DFF,
         eb2 + (size_t)l * D_MODEL, X, Tb, D_MODEL, D_MODEL, DFF, 0);
    lnorm(Tb, eln2_g + (size_t)l * D_MODEL, eln2_b + (size_t)l * D_MODEL, X);
    x = X;
  }
  lnorm(X, enc_norm_g, enc_norm_b, MEM);

  // ---------------- decoder ----------------
  const float* y = tgt;
  for (int l = 0; l < 6; ++l) {
    // local self-attention
    gemm(y, D_MODEL, dWqkv_sa + (size_t)l * 3 * D_MODEL * D_MODEL, D_MODEL,
         dbqkv_sa + (size_t)l * 3 * D_MODEL, nullptr, QKV, 3 * D_MODEL, 3 * D_MODEL, D_MODEL, 0);
    attn(QKV, 3 * D_MODEL, QKV + D_MODEL, QKV + 2 * D_MODEL, 3 * D_MODEL, ATTN, 1);
    gemm(ATTN, D_MODEL, dWo_sa + (size_t)l * D_MODEL * D_MODEL, D_MODEL,
         dbo_sa + (size_t)l * D_MODEL, y, Tb, D_MODEL, D_MODEL, D_MODEL, 0);
    lnorm(Tb, dln1_g + (size_t)l * D_MODEL, dln1_b + (size_t)l * D_MODEL, X);
    // cross-attention: Q from decoder state, K/V from encoder memory
    gemm(X, D_MODEL, dWqkv_ca + (size_t)l * 3 * D_MODEL * D_MODEL, D_MODEL,
         dbqkv_ca + (size_t)l * 3 * D_MODEL, nullptr, QKV, 3 * D_MODEL, D_MODEL, D_MODEL, 0);
    gemm(MEM, D_MODEL,
         dWqkv_ca + (size_t)l * 3 * D_MODEL * D_MODEL + (size_t)D_MODEL * D_MODEL, D_MODEL,
         dbqkv_ca + (size_t)l * 3 * D_MODEL + D_MODEL, nullptr,
         QKV + D_MODEL, 3 * D_MODEL, 2 * D_MODEL, D_MODEL, 0);
    attn(QKV, 3 * D_MODEL, QKV + D_MODEL, QKV + 2 * D_MODEL, 3 * D_MODEL, ATTN, 0);
    gemm(ATTN, D_MODEL, dWo_ca + (size_t)l * D_MODEL * D_MODEL, D_MODEL,
         dbo_ca + (size_t)l * D_MODEL, X, Tb, D_MODEL, D_MODEL, D_MODEL, 0);
    lnorm(Tb, dln2_g + (size_t)l * D_MODEL, dln2_b + (size_t)l * D_MODEL, X);
    // FFN
    gemm(X, D_MODEL, dW1 + (size_t)l * DFF * D_MODEL, D_MODEL,
         db1 + (size_t)l * DFF, nullptr, FF, DFF, DFF, D_MODEL, 1);
    gemm(FF, DFF, dW2 + (size_t)l * D_MODEL * DFF, DFF,
         db2 + (size_t)l * D_MODEL, X, Tb, D_MODEL, D_MODEL, DFF, 0);
    lnorm(Tb, dln3_g + (size_t)l * D_MODEL, dln3_b + (size_t)l * D_MODEL, X);
    y = X;
  }
  lnorm(X, dec_norm_g, dec_norm_b, (float*)d_out);
}
